// baseline_gae_30_80814104641732
// MI455X (gfx1250) — compile-verified, matrix operand order checked by reading, not run on hardware
//
#include <hip/hip_runtime.h>
#include <hip/hip_bf16.h>
#include <cstddef>

#define NTOT   15360
#define NEDGE  245760
#define NNODES 30
#define NBATCH 512

typedef __attribute__((ext_vector_type(16))) __bf16 v16bf;
typedef __attribute__((ext_vector_type(8)))  __bf16 v8bf;
typedef __attribute__((ext_vector_type(8)))  float  v8f;

enum { GF_BT = 1, GF_ACC = 2, GF_BIAS = 4, GF_RELU = 8 };

// ---------------------------------------------------------------- utilities
__global__ void zero4_kernel(float* p, int n4) {
  int i = blockIdx.x * blockDim.x + threadIdx.x;
  if (i < n4) ((float4*)p)[i] = make_float4(0.f, 0.f, 0.f, 0.f);
}

__global__ void deg_kernel(const int* __restrict__ src, float* deg, int E) {
  int e = blockIdx.x * blockDim.x + threadIdx.x;
  if (e < E) atomicAdd(deg + src[e], 1.0f);
}

__global__ void dinv_kernel(float* deg, int n) {
  int i = blockIdx.x * blockDim.x + threadIdx.x;
  if (i < n) {
    float d = deg[i];
    deg[i] = (d > 0.f) ? rsqrtf(fmaxf(d, 1.f)) : 0.f;
  }
}

__global__ void edgew_kernel(const int* __restrict__ src, const int* __restrict__ dst,
                             const float* __restrict__ dinv, float* w, int E) {
  int e = blockIdx.x * blockDim.x + threadIdx.x;
  if (e < E) w[e] = -dinv[src[e]] * dinv[dst[e]];
}

// out[dst] += w * h[src]   (float4-vectorized atomics)
__global__ void scatter_kernel(const int* __restrict__ src, const int* __restrict__ dst,
                               const float* __restrict__ w, const float* __restrict__ h,
                               float* out, int E, int F4) {
  int idx = blockIdx.x * blockDim.x + threadIdx.x;
  int total = E * F4;
  if (idx >= total) return;
  int e = idx / F4, g = idx % F4;
  float we = w[e];
  int s = src[e], d = dst[e];
  float4 hv = ((const float4*)(h + (size_t)s * F4 * 4))[g];
  float* o = out + (size_t)d * F4 * 4 + g * 4;
  atomicAdd(o + 0, we * hv.x);
  atomicAdd(o + 1, we * hv.y);
  atomicAdd(o + 2, we * hv.z);
  atomicAdd(o + 3, we * hv.w);
}

// p = 2*p - tx0
__global__ void cheb_comb_kernel(float* p, const float* __restrict__ tx0, int n4) {
  int i = blockIdx.x * blockDim.x + threadIdx.x;
  if (i < n4) {
    float4 a = ((float4*)p)[i];
    float4 b = ((const float4*)tx0)[i];
    ((float4*)p)[i] = make_float4(2.f * a.x - b.x, 2.f * a.y - b.y,
                                  2.f * a.z - b.z, 2.f * a.w - b.w);
  }
}

__global__ void add_kernel(const float* __restrict__ a, const float* __restrict__ b,
                           float* c, int n4) {
  int i = blockIdx.x * blockDim.x + threadIdx.x;
  if (i < n4) {
    float4 x = ((const float4*)a)[i];
    float4 y = ((const float4*)b)[i];
    ((float4*)c)[i] = make_float4(x.x + y.x, x.y + y.y, x.z + y.z, x.w + y.w);
  }
}

__global__ void gelu_kernel(float* p, int n) {
  int i = blockIdx.x * blockDim.x + threadIdx.x;
  if (i < n) {
    float x = p[i];
    p[i] = 0.5f * x * (1.f + erff(x * 0.70710678118654752f));
  }
}

// wave-per-row LayerNorm over 256 features, optional fused ReLU before LN
__global__ void ln256_kernel(const float* __restrict__ in, const float* __restrict__ g,
                             const float* __restrict__ b, float* out, int relu) {
  int tid = threadIdx.x;
  int lane = tid & 31, wave = tid >> 5;
  int row = blockIdx.x * 8 + wave;
  const float* r = in + (size_t)row * 256;
  float v[8], s = 0.f, s2 = 0.f;
#pragma unroll
  for (int j = 0; j < 8; j++) {
    float x = r[lane + 32 * j];
    if (relu) x = fmaxf(x, 0.f);
    v[j] = x; s += x; s2 += x * x;
  }
#pragma unroll
  for (int m = 16; m >= 1; m >>= 1) {
    s  += __shfl_xor(s,  m, 32);
    s2 += __shfl_xor(s2, m, 32);
  }
  float mean = s * (1.f / 256.f);
  float var  = s2 * (1.f / 256.f) - mean * mean;
  float rstd = rsqrtf(var + 1e-5f);
  float* o = out + (size_t)row * 256;
#pragma unroll
  for (int j = 0; j < 8; j++) {
    int f = lane + 32 * j;
    o[f] = (v[j] - mean) * rstd * g[f] + b[f];
  }
}

// -------------------------------------------------------- WMMA bf16 GEMM
// C[M,N] = act( [C +] A[M,K] @ B [+ bias] )
// BT=false: B is [K,N].  BT=true: B is [N,K] (weights stored [out,in]).
// Block: 256 thr = 8 waves. Block tile 128x64; wave tile 16x64 (4 WMMA
// accumulators sharing one A fragment). K-step 32, double-buffered LDS.
// All B-fragment LDS loads are hoisted ahead of the WMMA chain so the 4
// v_wmma issue back-to-back after a single s_wait_dscnt.
template <bool BT>
__global__ void gemm_wmma_kernel(const float* __restrict__ A, const float* __restrict__ B,
                                 float* __restrict__ C, const float* __restrict__ bias,
                                 int M, int N, int K, int flags) {
  // ldsB[buf][n][k], n-stride padded to 48 bf16 (96B) so each fragment half
  // is a 16B-aligned ds_load_b128.
  __shared__ __align__(16) __bf16 ldsB[2][64 * 48];

  int tid  = threadIdx.x;
  int lane = tid & 31, wave = tid >> 5;
  int khalf = lane >> 4;        // K-half select per ISA 16-bit A/B layout
  int nloc  = lane & 15;        // column within a 16-wide tile / A row
  int n0   = blockIdx.x * 64;
  int mrow = blockIdx.y * 128 + wave * 16 + nloc;
  const float* Arow = A + (size_t)mrow * K;

  // staging assignment: thread covers 8 k-consecutive elems of one column
  int snn = tid & 63;           // n within panel
  int skb = (tid >> 6) << 3;    // k base: 0,8,16,24

  auto stage = [&](int buf, int k0) {
    v8bf pk;
    if (BT) {
      const float* bsrc = B + (size_t)(n0 + snn) * K + (k0 + skb);
      float4 b0 = ((const float4*)bsrc)[0];
      float4 b1 = ((const float4*)bsrc)[1];
      pk[0]=(__bf16)b0.x; pk[1]=(__bf16)b0.y; pk[2]=(__bf16)b0.z; pk[3]=(__bf16)b0.w;
      pk[4]=(__bf16)b1.x; pk[5]=(__bf16)b1.y; pk[6]=(__bf16)b1.z; pk[7]=(__bf16)b1.w;
    } else {
      const float* bsrc = B + (size_t)(k0 + skb) * N + (n0 + snn);
#pragma unroll
      for (int j = 0; j < 8; j++) pk[j] = (__bf16)bsrc[(size_t)j * N];
    }
    *(v8bf*)&ldsB[buf][snn * 48 + skb] = pk;   // one 16B ds_store
  };

  v8f acc[4];
#pragma unroll
  for (int t = 0; t < 4; t++)
#pragma unroll
    for (int i = 0; i < 8; i++) acc[t][i] = 0.f;

  stage(0, 0);
  __syncthreads();

  int nsteps = K >> 5;
  int buf = 0;
  for (int s = 0; s < nsteps; s++) {
    int k0 = s << 5;
    if (s + 1 < nsteps) {
      stage(buf ^ 1, k0 + 32);                      // prefetch next panel
      __builtin_prefetch(Arow + k0 + 32, 0, 1);     // global_prefetch_b8
    }

    // A fragment: lane holds row m = lane&15, K-half = lane>>4
    const float* ap = Arow + k0 + 8 * khalf;
    float4 a0 = *(const float4*)(ap);
    float4 a1 = *(const float4*)(ap + 4);
    float4 a2 = *(const float4*)(ap + 16);
    float4 a3 = *(const float4*)(ap + 20);
    v16bf av;
    av[0]=(__bf16)a0.x;  av[1]=(__bf16)a0.y;  av[2]=(__bf16)a0.z;  av[3]=(__bf16)a0.w;
    av[4]=(__bf16)a1.x;  av[5]=(__bf16)a1.y;  av[6]=(__bf16)a1.z;  av[7]=(__bf16)a1.w;
    av[8]=(__bf16)a2.x;  av[9]=(__bf16)a2.y;  av[10]=(__bf16)a2.z; av[11]=(__bf16)a2.w;
    av[12]=(__bf16)a3.x; av[13]=(__bf16)a3.y; av[14]=(__bf16)a3.z; av[15]=(__bf16)a3.w;

    // hoist ALL B-fragment LDS loads (clause of 8 ds_load_b128, one wait)
    v8bf blo[4], bhi[4];
#pragma unroll
    for (int nt = 0; nt < 4; nt++) {
      const __bf16* bp = &ldsB[buf][(nt * 16 + nloc) * 48 + 8 * khalf];
      blo[nt] = *(const v8bf*)(bp);
      bhi[nt] = *(const v8bf*)(bp + 16);
    }

    // 4 WMMAs back-to-back, all reusing the same A fragment
#pragma unroll
    for (int nt = 0; nt < 4; nt++) {
      v16bf bv;
#pragma unroll
      for (int i = 0; i < 8; i++) { bv[i] = blo[nt][i]; bv[8 + i] = bhi[nt][i]; }
      acc[nt] = __builtin_amdgcn_wmma_f32_16x16x32_bf16(
          false, av, false, bv, (short)0, acc[nt], false, false);
    }
    __syncthreads();
    buf ^= 1;
  }

  // D layout: VGPR r holds M = r + 8*(lane>>4), N = lane&15
  int mbase = blockIdx.y * 128 + wave * 16 + 8 * khalf;
#pragma unroll
  for (int nt = 0; nt < 4; nt++) {
    int ng = n0 + nt * 16 + nloc;
    float bb = (flags & GF_BIAS) ? bias[ng] : 0.f;
#pragma unroll
    for (int r = 0; r < 8; r++) {
      size_t off = (size_t)(mbase + r) * N + ng;
      float val = acc[nt][r] + bb;
      if (flags & GF_ACC) val += C[off];
      if (flags & GF_RELU) val = fmaxf(val, 0.f);
      C[off] = val;
    }
  }
}

// ---------------------------------------------------------------- attention
// one block per (batch, head): S=30, Dh=64
__global__ void attn_kernel(const float* __restrict__ qkv, float* __restrict__ ao) {
  __shared__ float ks[NNODES * 64];
  __shared__ float vs[NNODES * 64];
  __shared__ float att[NNODES][31];
  int b = blockIdx.x >> 2;
  int head = blockIdx.x & 3;
  int base = b * NNODES;
  int tid = threadIdx.x;

  for (int i = tid; i < NNODES * 64; i += 256) {
    int s = i >> 6, d = i & 63;
    const float* row = qkv + (size_t)(base + s) * 768;
    ks[i] = row[256 + head * 64 + d];
    vs[i] = row[512 + head * 64 + d];
  }
  __syncthreads();

  for (int i = tid; i < NNODES * NNODES; i += 256) {
    int s = i / NNODES, t = i % NNODES;
    const float* q = qkv + (size_t)(base + s) * 768 + head * 64;
    float a = 0.f;
    for (int d = 0; d < 64; d++) a += q[d] * ks[t * 64 + d];
    att[s][t] = a * 0.125f;   // 1/sqrt(64)
  }
  __syncthreads();

  if (tid < NNODES) {
    float mx = -1e30f;
    for (int t = 0; t < NNODES; t++) mx = fmaxf(mx, att[tid][t]);
    float sum = 0.f;
    for (int t = 0; t < NNODES; t++) { float e = expf(att[tid][t] - mx); att[tid][t] = e; sum += e; }
    float inv = 1.f / sum;
    for (int t = 0; t < NNODES; t++) att[tid][t] *= inv;
  }
  __syncthreads();

  for (int i = tid; i < NNODES * 64; i += 256) {
    int s = i >> 6, d = i & 63;
    float a = 0.f;
    for (int t = 0; t < NNODES; t++) a += att[s][t] * vs[t * 64 + d];
    ao[(size_t)(base + s) * 256 + head * 64 + d] = a;
  }
}

// edge-feature gather: ef[b,e,:] = concat(h[b,si,:], h[b,di,:])
__global__ void gather_ef_kernel(const float* __restrict__ h, const int* __restrict__ pairs,
                                 float* __restrict__ ef) {
  int idx = blockIdx.x * blockDim.x + threadIdx.x;  // exactly NTOT*512 threads
  int row = idx >> 9, j = idx & 511;
  int bb = row / NNODES, e = row % NNODES;
  int node = (j < 256) ? pairs[2 * e] : pairs[2 * e + 1];
  ef[idx] = h[((size_t)(bb * NNODES + node)) * 256 + (j & 255)];
}

// wave-per-row final dot: out[r] = o3[r,:]·w4 + b4
__global__ void final_dot_kernel(const float* __restrict__ o3, const float* __restrict__ w4,
                                 const float* __restrict__ b4, float* out) {
  int tid = threadIdx.x;
  int lane = tid & 31, wave = tid >> 5;
  int row = blockIdx.x * 8 + wave;
  const float* r = o3 + (size_t)row * 128;
  float s = 0.f;
#pragma unroll
  for (int j = 0; j < 4; j++) { int f = lane + 32 * j; s += r[f] * w4[f]; }
#pragma unroll
  for (int m = 16; m >= 1; m >>= 1) s += __shfl_xor(s, m, 32);
  if (lane == 0) out[row] = s + b4[0];
}

// ---------------------------------------------------------------- launcher
extern "C" void kernel_launch(void* const* d_in, const int* in_sizes, int n_in,
                              void* d_out, int out_size, void* d_ws, size_t ws_size,
                              hipStream_t stream) {
  (void)in_sizes; (void)n_in; (void)out_size; (void)ws_size;
  const float* x      = (const float*)d_in[0];
  const int*   edge   = (const int*)d_in[1];
  const int*   src    = edge;
  const int*   dst    = edge + NEDGE;
  const int*   pairs  = (const int*)d_in[2];
  const float* chebW[3] = {(const float*)d_in[3], (const float*)d_in[5], (const float*)d_in[7]};
  const float* chebB[3] = {(const float*)d_in[4], (const float*)d_in[6], (const float*)d_in[8]};
  const float* normG[3] = {(const float*)d_in[9],  (const float*)d_in[11], (const float*)d_in[13]};
  const float* normB[3] = {(const float*)d_in[10], (const float*)d_in[12], (const float*)d_in[14]};
  const float* ln1_g = (const float*)d_in[15]; const float* ln1_b = (const float*)d_in[16];
  const float* ln2_g = (const float*)d_in[17]; const float* ln2_b = (const float*)d_in[18];
  const float* attn_in_w  = (const float*)d_in[19]; const float* attn_in_b  = (const float*)d_in[20];
  const float* attn_out_w = (const float*)d_in[21]; const float* attn_out_b = (const float*)d_in[22];
  const float* ffn_w1 = (const float*)d_in[23]; const float* ffn_b1 = (const float*)d_in[24];
  const float* ffn_w2 = (const float*)d_in[25]; const float* ffn_b2 = (const float*)d_in[26];
  const float* mlp_w1 = (const float*)d_in[27]; const float* mlp_b1 = (const float*)d_in[28];
  const float* mlp_w2 = (const float*)d_in[29]; const float* mlp_b2 = (const float*)d_in[30];
  const float* mlp_w3 = (const float*)d_in[31]; const float* mlp_b3 = (const float*)d_in[32];
  const float* mlp_w4 = (const float*)d_in[33]; const float* mlp_b4 = (const float*)d_in[34];

  float* ws   = (float*)d_ws;
  float* wE   = ws;                 // edge weights  [NEDGE]
  float* dinv = ws + NEDGE;         // deg -> dinv   [NTOT]
  const size_t NF = (size_t)NTOT * 256;
  float* R1 = dinv + NTOT;          // node features [NF]
  float* R2 = R1 + NF;
  float* R3 = R2 + NF;
  float* R4 = R3 + NF;
  float* R5 = R4 + NF;              // cheb accumulator / residual
  float* R6 = R5 + NF;              // wide buffer    [4*NF]  (qkv / ffn hidden)
  float* R7 = R6 + 4 * NF;          // edge features  [2*NF]
  float* R8 = R7 + 2 * NF;          // mlp hidden     [NF]

  auto zero = [&](float* p, size_t n) {
    zero4_kernel<<<(unsigned)(n / 1024), 256, 0, stream>>>(p, (int)(n / 4));
  };
  auto gemm = [&](const float* A, const float* B, float* C, const float* bias,
                  int M, int N, int K, int flags) {
    dim3 g((unsigned)(N / 64), (unsigned)(M / 128));
    if (flags & GF_BT)
      gemm_wmma_kernel<true><<<g, 256, 0, stream>>>(A, B, C, bias, M, N, K, flags);
    else
      gemm_wmma_kernel<false><<<g, 256, 0, stream>>>(A, B, C, bias, M, N, K, flags);
  };
  auto prop = [&](const float* h, float* out, int F) {
    zero(out, (size_t)NTOT * F);
    int total = NEDGE * (F / 4);
    scatter_kernel<<<total / 256, 256, 0, stream>>>(src, dst, wE, h, out, NEDGE, F / 4);
  };
  auto comb = [&](float* p, const float* tx0, int F) {
    cheb_comb_kernel<<<(unsigned)((size_t)NTOT * F / 1024), 256, 0, stream>>>(
        p, tx0, (int)((size_t)NTOT * F / 4));
  };

  // --- graph weights: deg -> dinv -> edge w
  zero(dinv, NTOT);
  deg_kernel<<<NEDGE / 256, 256, 0, stream>>>(src, dinv, NEDGE);
  dinv_kernel<<<NTOT / 256, 256, 0, stream>>>(dinv, NTOT);
  edgew_kernel<<<NEDGE / 256, 256, 0, stream>>>(src, dst, dinv, wE, NEDGE);

  // --- Cheb layer: acc = sum_k Tx_k @ W[k] + b ; h = LN(ReLU(acc))
  auto cheb_layer = [&](const float* hin, int Fin, const float* W, const float* bvec,
                        const float* g, const float* bb, float* hout) {
    size_t WS = (size_t)Fin * 256;
    gemm(hin, W + 0 * WS, R5, nullptr, NTOT, 256, Fin, 0);          // Tx0 @ W0
    prop(hin, R3, Fin);                                             // Tx1 = prop(Tx0)
    gemm(R3, W + 1 * WS, R5, nullptr, NTOT, 256, Fin, GF_ACC);
    prop(R3, R4, Fin); comb(R4, hin, Fin);                          // Tx2 = 2 prop(Tx1) - Tx0
    gemm(R4, W + 2 * WS, R5, nullptr, NTOT, 256, Fin, GF_ACC);
    prop(R4, R2, Fin); comb(R2, R3, Fin);                           // Tx3
    gemm(R2, W + 3 * WS, R5, nullptr, NTOT, 256, Fin, GF_ACC);
    prop(R2, R3, Fin); comb(R3, R4, Fin);                           // Tx4
    gemm(R3, W + 4 * WS, R5, bvec, NTOT, 256, Fin, GF_ACC | GF_BIAS);
    ln256_kernel<<<NTOT / 8, 256, 0, stream>>>(R5, g, bb, hout, 1); // LN(ReLU(.))
  };

  cheb_layer(x,  128, chebW[0], chebB[0], normG[0], normB[0], R1);
  cheb_layer(R1, 256, chebW[1], chebB[1], normG[1], normB[1], R1);
  cheb_layer(R1, 256, chebW[2], chebB[2], normG[2], normB[2], R1);

  // --- transformer block (h3 in R1)
  ln256_kernel<<<NTOT / 8, 256, 0, stream>>>(R1, ln1_g, ln1_b, R2, 0);
  gemm(R2, attn_in_w, R6, attn_in_b, NTOT, 768, 256, GF_BT | GF_BIAS);  // qkv
  attn_kernel<<<NBATCH * 4, 256, 0, stream>>>(R6, R3);                  // ao
  gemm(R3, attn_out_w, R4, attn_out_b, NTOT, 256, 256, GF_BT | GF_BIAS);
  add_kernel<<<(unsigned)(NF / 1024), 256, 0, stream>>>(R1, R4, R5, (int)(NF / 4)); // h4
  ln256_kernel<<<NTOT / 8, 256, 0, stream>>>(R5, ln2_g, ln2_b, R2, 0);
  gemm(R2, ffn_w1, R6, ffn_b1, NTOT, 1024, 256, GF_BT | GF_BIAS);
  gelu_kernel<<<NTOT * 1024 / 256, 256, 0, stream>>>(R6, NTOT * 1024);
  gemm(R6, ffn_w2, R3, ffn_b2, NTOT, 256, 1024, GF_BT | GF_BIAS);
  add_kernel<<<(unsigned)(NF / 1024), 256, 0, stream>>>(R5, R3, R1, (int)(NF / 4)); // h6

  // --- edge MLP
  gather_ef_kernel<<<NTOT * 512 / 256, 256, 0, stream>>>(R1, pairs, R7);
  gemm(R7, mlp_w1, R8, mlp_b1, NTOT, 256, 512, GF_BT | GF_BIAS | GF_RELU);
  gemm(R8, mlp_w2, R2, mlp_b2, NTOT, 128, 256, GF_BT | GF_BIAS | GF_RELU);
  gemm(R2, mlp_w3, R3, mlp_b3, NTOT, 128, 128, GF_BT | GF_BIAS | GF_RELU);
  final_dot_kernel<<<NTOT / 8, 256, 0, stream>>>(R3, mlp_w4, mlp_b4, (float*)d_out);
}
